// StreamingAttention_7043746365912
// MI455X (gfx1250) — matrix operand order repaired and checked by reading, not compiled, not run on hardware
//
#include <hip/hip_runtime.h>
#include <cstdint>

#define DEVINL __device__ __forceinline__

typedef __attribute__((ext_vector_type(16))) __bf16 v16bf;
typedef __attribute__((ext_vector_type(8)))  float  v8f;
typedef __attribute__((ext_vector_type(4)))  unsigned int u32x4;
typedef __attribute__((ext_vector_type(4)))  int   i32x4;
typedef __attribute__((ext_vector_type(8)))  int   i32x8;

constexpr int Bb = 4, Ss = 4096, Ee = 1024, Hh = 16, Dd = 64;
constexpr int Cc = 128, Oo = 32;
constexpr int NCHUNK = Ss / Cc;                        // 32
constexpr int SOUT   = Cc + (NCHUNK - 1) * (Cc - Oo);  // 3104
constexpr float SCALE = 0.125f;                        // 1/sqrt(64)

union FragU { v16bf v; uint4 q[2]; };

DEVINL v8f wmma_bf16(v16bf a, v16bf b, v8f c) {
  return __builtin_amdgcn_wmma_f32_16x16x32_bf16(false, a, false, b, (short)0, c,
                                                 false, false);
}

DEVINL v8f zero8() {
  v8f z = {0.f, 0.f, 0.f, 0.f, 0.f, 0.f, 0.f, 0.f};
  return z;
}

// ---------------------------------------------------------------------------
// Tensor Data Mover: async 2D bf16 tile load (global -> LDS).
// D# per CDNA5 ISA: group0 = {flags, lds_addr,
// global_addr[56:0], type=2}; group1 = {data_size=2B, tensor_dim0/1,
// tile_dim0/1, tensor_dim0_stride}; groups 2/3 zero (<=2D tensor).
// Issue from ONE wave only; completion via TENSORcnt + workgroup barrier.
// ---------------------------------------------------------------------------
DEVINL void tdm_load_2d(uint32_t lds_byte_off, const void* gptr,
                        uint32_t rows, uint32_t row_elems,
                        uint32_t row_stride_elems) {
  uint64_t ga = (uint64_t)(uintptr_t)gptr;
  u32x4 g0;
  g0[0] = 1u;                                               // count=1 (valid)
  g0[1] = lds_byte_off;                                     // lds_addr
  g0[2] = (uint32_t)ga;                                     // global_addr[31:0]
  g0[3] = (uint32_t)((ga >> 32) & 0x01FFFFFFu) | (2u << 30);// ga[56:32] | type=2
  i32x8 g1;
  g1[0] = (int)(1u << 16);                                  // data_size=1 (2B)
  g1[1] = (int)((row_elems & 0xFFFFu) << 16);               // tensor_dim0[15:0]
  g1[2] = (int)(((row_elems >> 16) & 0xFFFFu) |
                ((rows & 0xFFFFu) << 16));                  // td0[31:16]|td1[15:0]
  g1[3] = (int)(((rows >> 16) & 0xFFFFu) |
                ((row_elems & 0xFFFFu) << 16));             // td1[31:16]|tile_dim0
  g1[4] = (int)(rows & 0xFFFFu);                            // tile_dim1 (tile_dim2=0)
  g1[5] = (int)row_stride_elems;                            // tensor_dim0_stride[31:0]
  g1[6] = 0;                                                // stride[47:32]
  g1[7] = 0;
  i32x4 z4 = {0, 0, 0, 0};
#if defined(__clang_major__) && (__clang_major__ >= 23)
  i32x8 z8 = {0, 0, 0, 0, 0, 0, 0, 0};
  __builtin_amdgcn_tensor_load_to_lds(g0, g1, z4, z4, z8, 0);
#else
  __builtin_amdgcn_tensor_load_to_lds(g0, g1, z4, z4, 0);
#endif
}

// A-fragment (16x32 bf16) from row-major [m][k], stride `ld` elements.
DEVINL v16bf load_a(const __bf16* base, int m0, int ld, int k0, int lane) {
  int m  = m0 + (lane & 15);
  int kb = k0 + ((lane < 16) ? 0 : 8);
  const __bf16* p = base + (size_t)m * ld + kb;
  FragU f;
  f.q[0] = *(const uint4*)(p);
  f.q[1] = *(const uint4*)(p + 16);
  return f.v;
}

// B-fragment (32x16 bf16) from N-major [n][k], stride `ld` elements.
DEVINL v16bf load_b(const __bf16* base, int n0, int ld, int k0, int lane) {
  int n  = n0 + (lane & 15);
  int kh = k0 + ((lane < 16) ? 0 : 8);
  const __bf16* p = base + (size_t)n * ld + kh;
  FragU f;
  f.q[0] = *(const uint4*)(p);
  f.q[1] = *(const uint4*)(p + 16);
  return f.v;
}

// ---------------------------------------------------------------------------
// One-time weight prep: Wt[n][k] = (bf16)W[k][n]   (1024x1024, LDS-tiled)
// ---------------------------------------------------------------------------
__global__ __launch_bounds__(256) void transpose_w(const float* __restrict__ W,
                                                   __bf16* __restrict__ Wt) {
  __shared__ __bf16 t[64][65];
  const int n0 = blockIdx.x * 64, k0 = blockIdx.y * 64;
  for (int i = threadIdx.x; i < 64 * 64; i += 256) {
    int r = i >> 6, c = i & 63;  // r: k, c: n  (coalesced read over n)
    t[c][r] = (__bf16)W[(size_t)(k0 + r) * 1024 + n0 + c];
  }
  __syncthreads();
  for (int i = threadIdx.x; i < 64 * 64; i += 256) {
    int r = i >> 6, c = i & 63;  // r: n, c: k  (coalesced write over k)
    Wt[(size_t)(n0 + r) * 1024 + k0 + c] = t[r][c];
  }
}

// ---------------------------------------------------------------------------
// GEMM: Out[M x 1024] = A[M x 1024] * W + bias, W pre-transposed bf16 [n][k].
// Block tile 128x128, BK=32, 8 waves, wave tile 64x32.
// W tiles (and A tiles when bf16) staged by TDM, double-buffered.
// ---------------------------------------------------------------------------
template <bool A_F32, bool OUT_F32>
__global__ __launch_bounds__(256) void gemm_bias(const void* __restrict__ Ap,
                                                 const __bf16* __restrict__ Wt,
                                                 const float* __restrict__ bias,
                                                 void* __restrict__ Outp) {
  constexpr int NIT = 1024 / 32;
  // LDS layout (bytes): A_F32: sA@0 (8K), sB0@8K, sB1@16K
  //                     !A_F32: sA0@0, sA1@8K, sB0@16K, sB1@24K
  __shared__ __align__(16) char smem[A_F32 ? 24576 : 32768];
  constexpr uint32_t offA0 = 0, offA1 = 8192;
  constexpr uint32_t offB0 = A_F32 ? 8192 : 16384;
  constexpr uint32_t offB1 = offB0 + 8192;

  const int tid  = threadIdx.x;
  const int lane = tid & 31;
  const int w    = tid >> 5;
  const int m0w  = (w >> 2) * 64;
  const int n0w  = (w & 3) * 32;
  const size_t rowBase = (size_t)blockIdx.y * 128;
  const int    colBase = blockIdx.x * 128;

  const float*  Af  = (const float*)Ap;
  const __bf16* Abp = (const __bf16*)Ap;

  // prologue: async-stage tiles for it=0 (wave 0 only; TDM ignores EXEC)
  if (w == 0) {
    tdm_load_2d(offB0, Wt + (size_t)colBase * 1024, 128, 32, 1024);
    if constexpr (!A_F32)
      tdm_load_2d(offA0, Abp + rowBase * 1024, 128, 32, 1024);
  }

  v8f acc[4][2];
#pragma unroll
  for (int mt = 0; mt < 4; ++mt)
#pragma unroll
    for (int nt = 0; nt < 2; ++nt) acc[mt][nt] = zero8();

  for (int it = 0; it < NIT; ++it) {
    const int k0 = it * 32;
    if constexpr (A_F32) {
      // manual f32 -> bf16 stage of A tile into sA
      __bf16* sA = (__bf16*)smem;
#pragma unroll
      for (int c = 0; c < 4; ++c) {
        int chunk = c * 256 + tid;  // 1024 chunks of 4 floats
        int r = chunk >> 3, c4 = (chunk & 7) * 4;
        float4 v = *(const float4*)(Af + (rowBase + r) * 1024 + k0 + c4);
        union { __bf16 h4[4]; uint2 u; } cv;
        cv.h4[0] = (__bf16)v.x; cv.h4[1] = (__bf16)v.y;
        cv.h4[2] = (__bf16)v.z; cv.h4[3] = (__bf16)v.w;
        *(uint2*)&sA[r * 32 + c4] = cv.u;
      }
    }
    // prefetch next K-step via TDM; wait for current tiles
    if (w == 0) {
      if (it + 1 < NIT) {
        uint32_t bb = ((it + 1) & 1) ? offB1 : offB0;
        tdm_load_2d(bb, Wt + (size_t)colBase * 1024 + (k0 + 32), 128, 32, 1024);
        if constexpr (!A_F32) {
          uint32_t ab = ((it + 1) & 1) ? offA1 : offA0;
          tdm_load_2d(ab, Abp + rowBase * 1024 + (k0 + 32), 128, 32, 1024);
          __builtin_amdgcn_s_wait_tensorcnt(2);
        } else {
          __builtin_amdgcn_s_wait_tensorcnt(1);
        }
      } else {
        __builtin_amdgcn_s_wait_tensorcnt(0);
      }
    }
    __syncthreads();

    const __bf16* curA =
        A_F32 ? (const __bf16*)smem
              : (const __bf16*)(smem + ((it & 1) ? offA1 : offA0));
    const __bf16* curB = (const __bf16*)(smem + ((it & 1) ? offB1 : offB0));

    v16bf af[4];
#pragma unroll
    for (int mt = 0; mt < 4; ++mt) af[mt] = load_a(curA, m0w + mt * 16, 32, 0, lane);
    v16bf bf2[2];
#pragma unroll
    for (int nt = 0; nt < 2; ++nt) bf2[nt] = load_b(curB, n0w + nt * 16, 32, 0, lane);
#pragma unroll
    for (int mt = 0; mt < 4; ++mt)
#pragma unroll
      for (int nt = 0; nt < 2; ++nt)
        acc[mt][nt] = wmma_bf16(af[mt], bf2[nt], acc[mt][nt]);
    __syncthreads();
  }

  const int hi = lane >> 4, lc = lane & 15;
#pragma unroll
  for (int mt = 0; mt < 4; ++mt)
#pragma unroll
    for (int nt = 0; nt < 2; ++nt)
#pragma unroll
      for (int i = 0; i < 8; ++i) {
        size_t r = rowBase + m0w + mt * 16 + i + hi * 8;
        int    c = colBase + n0w + nt * 16 + lc;
        float  v = acc[mt][nt][i] + bias[c];
        if constexpr (OUT_F32)
          ((float*)Outp)[r * 1024 + c] = v;
        else
          ((__bf16*)Outp)[r * 1024 + c] = (__bf16)v;
      }
}

// ---------------------------------------------------------------------------
// Streaming attention per (head, chunk, batch). Q/K staged by TDM; V staged
// transposed manually; scale folded into softmax.
// ---------------------------------------------------------------------------
template <bool FIRST>
__global__ __launch_bounds__(256) void attn_kernel(const __bf16* __restrict__ Q,
                                                   const __bf16* __restrict__ K,
                                                   const __bf16* __restrict__ V,
                                                   __bf16* __restrict__ Oa) {
  constexpr int nQ  = FIRST ? 128 : 96;
  constexpr int nK  = FIRST ? 128 : 160;
  constexpr int NKT = nK / 16;
  constexpr int NKS = nK / 32;

  const int h     = blockIdx.x;
  const int chunk = FIRST ? 0 : (int)blockIdx.y + 1;
  const int b     = blockIdx.z;
  const int qbase = FIRST ? 0 : chunk * Cc + Oo;
  const int kb0   = FIRST ? 0 : chunk * Cc - Oo;
  const int obase = FIRST ? 0 : Cc + (chunk - 1) * (Cc - Oo);

  const int tid = threadIdx.x, lane = tid & 31, w = tid >> 5;

  // LDS (bytes): sQ@0 [128][64], sK@16384 [160][64], sVt@36864 [64][160],
  //              sP@57344 [128][160]
  __shared__ __align__(16) char smem[98304];
  __bf16* sQ  = (__bf16*)(smem);
  __bf16* sK  = (__bf16*)(smem + 16384);
  __bf16* sVt = (__bf16*)(smem + 36864);
  __bf16* sP  = (__bf16*)(smem + 57344);

  const __bf16* qg = Q + ((size_t)b * Ss + qbase) * Ee + h * Dd;
  const __bf16* kg = K + ((size_t)b * Ss + kb0) * Ee + h * Dd;
  const __bf16* vg = V + ((size_t)b * Ss + kb0) * Ee + h * Dd;

  // async stage Q and K tiles (wave 0)
  if (w == 0) {
    tdm_load_2d(0u, qg, nQ, 64, 1024);
    tdm_load_2d(16384u, kg, nK, 64, 1024);
  }
  // stage V transposed (overlaps with TDM)
  for (int i = tid; i < nK * 64; i += 256) {
    int r = i >> 6, d = i & 63;
    sVt[d * 160 + r] = vg[(size_t)r * Ee + d];
  }
  // zero-pad unused query rows (disjoint from TDM destination rows)
  if constexpr (!FIRST) {
    for (int i = tid; i < (128 - nQ) * 16; i += 256) {
      int r = nQ + (i >> 4), c4 = (i & 15) * 4;
      uint2 z = make_uint2(0u, 0u);
      *(uint2*)&sQ[r * 64 + c4] = z;
    }
  }
  if (w == 0) __builtin_amdgcn_s_wait_tensorcnt(0);
  __syncthreads();

  const int m0 = w * 16;

  // scores S = Q.K^T (f32 accum in registers)
  v8f sacc[NKT];
#pragma unroll
  for (int t = 0; t < NKT; ++t) sacc[t] = zero8();
#pragma unroll
  for (int ks = 0; ks < 64; ks += 32) {
    v16bf aq = load_a(sQ, m0, 64, ks, lane);
#pragma unroll
    for (int t = 0; t < NKT; ++t) {
      v16bf bk = load_b(sK, t * 16, 64, ks, lane);
      sacc[t] = wmma_bf16(aq, bk, sacc[t]);
    }
  }

  // fold in 1/sqrt(D), then softmax per row (rows live in 16-lane halves)
#pragma unroll
  for (int t = 0; t < NKT; ++t) sacc[t] = sacc[t] * SCALE;

  const int hi = lane >> 4, lc = lane & 15;
#pragma unroll
  for (int i = 0; i < 8; ++i) {
    float mx = -3.0e38f;
#pragma unroll
    for (int t = 0; t < NKT; ++t) mx = fmaxf(mx, sacc[t][i]);
#pragma unroll
    for (int msk = 1; msk < 16; msk <<= 1) mx = fmaxf(mx, __shfl_xor(mx, msk, 32));
    float sum = 0.f;
#pragma unroll
    for (int t = 0; t < NKT; ++t) {
      float p = __expf(sacc[t][i] - mx);
      sacc[t][i] = p;
      sum += p;
    }
#pragma unroll
    for (int msk = 1; msk < 16; msk <<= 1) sum += __shfl_xor(sum, msk, 32);
    float inv = 1.f / sum;
    int r = m0 + i + hi * 8;
#pragma unroll
    for (int t = 0; t < NKT; ++t) sP[r * 160 + t * 16 + lc] = (__bf16)(sacc[t][i] * inv);
  }
  __syncthreads();

  // O = P.V
  v8f oacc[4];
#pragma unroll
  for (int nt = 0; nt < 4; ++nt) oacc[nt] = zero8();
#pragma unroll
  for (int s = 0; s < NKS; ++s) {
    v16bf ap = load_a(sP, m0, 160, s * 32, lane);
#pragma unroll
    for (int nt = 0; nt < 4; ++nt) {
      v16bf bv = load_b(sVt, nt * 16, 160, s * 32, lane);
      oacc[nt] = wmma_bf16(ap, bv, oacc[nt]);
    }
  }

  __bf16* og = Oa + ((size_t)b * SOUT + obase) * Ee + h * Dd;
#pragma unroll
  for (int nt = 0; nt < 4; ++nt)
#pragma unroll
    for (int i = 0; i < 8; ++i) {
      int r = m0 + i + hi * 8;
      if (r < nQ) og[(size_t)r * Ee + nt * 16 + lc] = (__bf16)oacc[nt][i];
    }
}

// ---------------------------------------------------------------------------
extern "C" void kernel_launch(void* const* d_in, const int* in_sizes, int n_in,
                              void* d_out, int out_size, void* d_ws, size_t ws_size,
                              hipStream_t stream) {
  (void)in_sizes; (void)n_in; (void)out_size; (void)ws_size;

  const float* query  = (const float*)d_in[0];
  const float* key_in = (const float*)d_in[1];
  const float* value  = (const float*)d_in[2];
  const float* Wq = (const float*)d_in[3];
  const float* bq = (const float*)d_in[4];
  const float* Wk = (const float*)d_in[5];
  const float* bk = (const float*)d_in[6];
  const float* Wv = (const float*)d_in[7];
  const float* bv = (const float*)d_in[8];
  const float* Wo = (const float*)d_in[9];
  const float* bo = (const float*)d_in[10];

  // workspace: Q,K,V bf16 [B,S,E]; attn out bf16 [B,SOUT,E]; 4 bf16 Wt [n][k]
  const size_t MS = (size_t)Bb * Ss * Ee;
  const size_t WS = (size_t)Ee * Ee;
  __bf16* Qb  = (__bf16*)d_ws;
  __bf16* Kb  = Qb + MS;
  __bf16* Vb  = Kb + MS;
  __bf16* Ab  = Vb + MS;
  __bf16* Wtq = Ab + (size_t)Bb * SOUT * Ee;
  __bf16* Wtk = Wtq + WS;
  __bf16* Wtv = Wtk + WS;
  __bf16* Wto = Wtv + WS;

  dim3 blk(256);

  // one-time weight convert+transpose (24 MB total traffic)
  dim3 gt(16, 16);
  transpose_w<<<gt, blk, 0, stream>>>(Wq, Wtq);
  transpose_w<<<gt, blk, 0, stream>>>(Wk, Wtk);
  transpose_w<<<gt, blk, 0, stream>>>(Wv, Wtv);
  transpose_w<<<gt, blk, 0, stream>>>(Wo, Wto);

  // QKV projections: M = B*S = 16384
  dim3 g1(Ee / 128, (Bb * Ss) / 128);
  gemm_bias<true, false><<<g1, blk, 0, stream>>>(query,  Wtq, bq, Qb);
  gemm_bias<true, false><<<g1, blk, 0, stream>>>(key_in, Wtk, bk, Kb);
  gemm_bias<true, false><<<g1, blk, 0, stream>>>(value,  Wtv, bv, Vb);

  // attention: chunk 0, then streaming chunks 1..31
  attn_kernel<true ><<<dim3(Hh, 1,          Bb), blk, 0, stream>>>(Qb, Kb, Vb, Ab);
  attn_kernel<false><<<dim3(Hh, NCHUNK - 1, Bb), blk, 0, stream>>>(Qb, Kb, Vb, Ab);

  // output projection: M = B*SOUT = 12416
  dim3 g2(Ee / 128, (Bb * SOUT) / 128);
  gemm_bias<false, true><<<g2, blk, 0, stream>>>(Ab, Wto, bo, d_out);
}
